// Bengio03HighwayBiLm_62689342652477
// MI455X (gfx1250) — matrix operand order repaired and checked by reading, not compiled, not run on hardware
//
#include <hip/hip_runtime.h>
#include <math.h>

// Problem constants (match reference)
#define B_   32
#define S_   512
#define H_   512
#define L_   2
#define NHW_ 2
#define W_   3
#define IN_  2048            // H*(W+1)
#define SP_  (S_ + 2*W_)     // padded sequence length = 518

typedef __bf16 bf16_t;
typedef bf16_t v16bf __attribute__((ext_vector_type(16)));
typedef bf16_t v8bf  __attribute__((ext_vector_type(8)));
typedef float  v8f   __attribute__((ext_vector_type(8)));
typedef float  v4f   __attribute__((ext_vector_type(4)));

#define BK  32               // K-step per LDS stage (one wmma K)
#define BKP 40               // padded LDS pitch (bf16 elems): 80B rows -> conflict-free, 16B aligned

static __device__ __forceinline__ v16bf cat8(v8bf lo, v8bf hi) {
  return __builtin_shufflevector(lo, hi, 0,1,2,3,4,5,6,7,8,9,10,11,12,13,14,15);
}

static __device__ __forceinline__ v8f wmma_bf16(v16bf a, v16bf b, v8f c) {
  // v_wmma_f32_16x16x32_bf16: D = A*B + C, f32 accumulate
  return __builtin_amdgcn_wmma_f32_16x16x32_bf16(false, a, false, b, (short)0, c, false, false);
}

// ---------------------------------------------------------------------------
// f32 -> bf16 weight conversion
// ---------------------------------------------------------------------------
__global__ void cvt_bf16_kernel(const float* __restrict__ src, bf16_t* __restrict__ dst, long n) {
  long i = (long)blockIdx.x * blockDim.x + threadIdx.x;
  if (i < n) dst[i] = (bf16_t)src[i];
}

// ---------------------------------------------------------------------------
// Build padded sequence: [B, SP, H] = [pad_f (W rows) | x (S rows) | pad_b (W rows)]
// ---------------------------------------------------------------------------
__global__ void build_padded_kernel(const float* __restrict__ x,
                                    const float* __restrict__ padF,
                                    const float* __restrict__ padB,
                                    float* __restrict__ out) {
  long i = (long)blockIdx.x * blockDim.x + threadIdx.x;
  if (i >= (long)B_ * SP_ * H_) return;
  int c  = (int)(i % H_);
  long t = i / H_;
  int tp = (int)(t % SP_);
  int b  = (int)(t / SP_);
  float v;
  if (tp < W_)            v = padF[tp * H_ + c];
  else if (tp < W_ + S_)  v = x[((long)b * S_ + (tp - W_)) * H_ + c];
  else                    v = padB[(tp - W_ - S_) * H_ + c];
  out[i] = v;
}

// ---------------------------------------------------------------------------
// Conv-as-GEMM + bias + ReLU, double-buffered LDS (1 barrier / K-step).
// C[z][m][n] = relu( sum_k A[z][m*lda + k] * Bw[n*ldb + k] + bias[n] )
// A rows overlap (lda = H < K): the 4-tap window over the padded sequence.
// Block tile 128x128, 8 waves (4x2), wave tile 32x64 (2x4 wmma tiles).
// ---------------------------------------------------------------------------
__global__ __launch_bounds__(256) void conv_gemm_relu_kernel(
    const float*  __restrict__ A,  long aZStride, long aOff, int lda,
    const bf16_t* __restrict__ Bw, int ldb,
    const float*  __restrict__ bias,
    float*        __restrict__ C,  long cZStride, int ldc,
    int K)
{
  __shared__ __align__(16) bf16_t sA[2][128][BKP];   // 2 x 10KB
  __shared__ __align__(16) bf16_t sB[2][128][BKP];   // 2 x 10KB

  const int tid  = threadIdx.x;
  const int lane = tid & 31;
  const int wave = tid >> 5;
  const int wm   = wave >> 1;   // 0..3  (wave row of 32)
  const int wn   = wave & 1;    // 0..1  (wave col of 64)
  const int lh   = lane & 15;
  const int ls   = lane >> 4;

  const int m0 = blockIdx.y * 128;
  const int n0 = blockIdx.x * 128;
  const float*  Ab = A + (long)blockIdx.z * aZStride + aOff + (long)m0 * lda;
  const bf16_t* Bb = Bw + (long)n0 * ldb;
  float*        Cb = C + (long)blockIdx.z * cZStride + (long)m0 * ldc;

  v8f acc[2][4] = {};

  // Register staging for global->LDS (overlaps global latency with compute)
  v4f  aReg[4];
  v8bf bReg[2];

  auto loadG = [&](int kk) {
    #pragma unroll
    for (int t = 0; t < 4; ++t) {
      int c = tid + t * 256, row = c >> 3, q = c & 7;
      aReg[t] = *(const v4f*)(Ab + (long)row * lda + kk + q * 4);
    }
    #pragma unroll
    for (int t = 0; t < 2; ++t) {
      int c = tid + t * 256, row = c >> 2, q = c & 3;
      bReg[t] = *(const v8bf*)(Bb + (long)row * ldb + kk + q * 8);
    }
  };
  auto storeS = [&](int buf) {
    #pragma unroll
    for (int t = 0; t < 4; ++t) {
      int c = tid + t * 256, row = c >> 3, q = c & 7;
      bf16_t* d = &sA[buf][row][q * 4];
      d[0] = (bf16_t)aReg[t].x; d[1] = (bf16_t)aReg[t].y;
      d[2] = (bf16_t)aReg[t].z; d[3] = (bf16_t)aReg[t].w;
    }
    #pragma unroll
    for (int t = 0; t < 2; ++t) {
      int c = tid + t * 256, row = c >> 2, q = c & 3;
      *(v8bf*)(&sB[buf][row][q * 8]) = bReg[t];
    }
  };

  const int NK = K / BK;
  loadG(0);
  storeS(0);
  __syncthreads();

  int buf = 0;
  for (int kt = 0; kt < NK; ++kt) {
    if (kt + 1 < NK) loadG((kt + 1) * BK);   // prefetch next tile into regs

    // A fragments (contiguous K order: lane half selects K 0..15 / 16..31)
    v16bf af[2];
    #pragma unroll
    for (int mt = 0; mt < 2; ++mt) {
      const bf16_t* ar = &sA[buf][wm * 32 + mt * 16 + lh][ls * 16];
      af[mt] = cat8(*(const v8bf*)ar, *(const v8bf*)(ar + 8));
    }
    #pragma unroll
    for (int nt = 0; nt < 4; ++nt) {
      // B fragment: lane = N column, lane-half selects K 0..15 / 16..31
      const bf16_t* br = &sB[buf][wn * 64 + nt * 16 + lh][ls * 16];
      v16bf bfrag = cat8(*(const v8bf*)br, *(const v8bf*)(br + 8));
      #pragma unroll
      for (int mt = 0; mt < 2; ++mt)
        acc[mt][nt] = wmma_bf16(af[mt], bfrag, acc[mt][nt]);
    }

    if (kt + 1 < NK) {
      storeS(buf ^ 1);        // safe: no thread can still be reading buf^1
      __syncthreads();
      buf ^= 1;
    }
  }

  // Epilogue: C layout — VGPR r holds M = r + 8*(lane>=16), N = lane&15
  #pragma unroll
  for (int nt = 0; nt < 4; ++nt) {
    int n = n0 + wn * 64 + nt * 16 + lh;
    float bn = bias[n];
    #pragma unroll
    for (int mt = 0; mt < 2; ++mt) {
      #pragma unroll
      for (int r = 0; r < 8; ++r) {
        int mloc = wm * 32 + mt * 16 + r + ls * 8;
        float v = acc[mt][nt][r] + bn;
        Cb[(long)mloc * ldc + n] = fmaxf(v, 0.0f);
      }
    }
  }
}

// ---------------------------------------------------------------------------
// Fused highway stage, double-buffered LDS:
//   nonlin = X @ Wn[0:H]^T + bias[0:H]
//   gate   = X @ Wn[H:2H]^T + bias[H:2H]
//   y      = sigmoid(gate) * X + (1 - sigmoid(gate)) * relu(nonlin)
// Block tile 128x64 (per half), 8 waves (4x2), wave tile 32x32, two acc sets.
// Optional second store (strided) writes the [L,B,S,2H] concat output.
// ---------------------------------------------------------------------------
__global__ __launch_bounds__(256) void highway_gemm_kernel(
    const float*  __restrict__ X,    // [M, H] — GEMM A and residual x_old
    const bf16_t* __restrict__ Wn,   // [2H, H] bf16
    const float*  __restrict__ bias, // [2H]
    float*        __restrict__ Out1, // [M, H]
    float*        __restrict__ Out2, // optional d_out slice, row stride ldo2
    int ldo2)
{
  const int K = H_;
  __shared__ __align__(16) bf16_t sA [2][128][BKP];
  __shared__ __align__(16) bf16_t sBn[2][64][BKP];
  __shared__ __align__(16) bf16_t sBg[2][64][BKP];

  const int tid  = threadIdx.x;
  const int lane = tid & 31;
  const int wave = tid >> 5;
  const int wm   = wave >> 1;   // 0..3
  const int wn   = wave & 1;    // 0..1
  const int lh   = lane & 15;
  const int ls   = lane >> 4;

  const int m0 = blockIdx.y * 128;
  const int n0 = blockIdx.x * 64;
  const float* Ab = X + (long)m0 * K;

  v8f accN[2][2] = {};
  v8f accG[2][2] = {};

  v4f  aReg[4];
  v8bf bnReg, bgReg;

  auto loadG = [&](int kk) {
    #pragma unroll
    for (int t = 0; t < 4; ++t) {
      int c = tid + t * 256, row = c >> 3, q = c & 7;
      aReg[t] = *(const v4f*)(Ab + (long)row * K + kk + q * 4);
    }
    {
      int row = tid >> 2, q = tid & 3;
      bnReg = *(const v8bf*)(Wn + (long)(n0 + row) * K + kk + q * 8);
      bgReg = *(const v8bf*)(Wn + (long)(H_ + n0 + row) * K + kk + q * 8);
    }
  };
  auto storeS = [&](int buf) {
    #pragma unroll
    for (int t = 0; t < 4; ++t) {
      int c = tid + t * 256, row = c >> 3, q = c & 7;
      bf16_t* d = &sA[buf][row][q * 4];
      d[0] = (bf16_t)aReg[t].x; d[1] = (bf16_t)aReg[t].y;
      d[2] = (bf16_t)aReg[t].z; d[3] = (bf16_t)aReg[t].w;
    }
    {
      int row = tid >> 2, q = tid & 3;
      *(v8bf*)(&sBn[buf][row][q * 8]) = bnReg;
      *(v8bf*)(&sBg[buf][row][q * 8]) = bgReg;
    }
  };

  const int NK = K / BK;   // 16
  loadG(0);
  storeS(0);
  __syncthreads();

  int buf = 0;
  for (int kt = 0; kt < NK; ++kt) {
    if (kt + 1 < NK) loadG((kt + 1) * BK);

    v16bf af[2];
    #pragma unroll
    for (int mt = 0; mt < 2; ++mt) {
      const bf16_t* ar = &sA[buf][wm * 32 + mt * 16 + lh][ls * 16];
      af[mt] = cat8(*(const v8bf*)ar, *(const v8bf*)(ar + 8));
    }
    #pragma unroll
    for (int nt = 0; nt < 2; ++nt) {
      const bf16_t* brn = &sBn[buf][wn * 32 + nt * 16 + lh][ls * 16];
      const bf16_t* brg = &sBg[buf][wn * 32 + nt * 16 + lh][ls * 16];
      v16bf bn = cat8(*(const v8bf*)brn, *(const v8bf*)(brn + 8));
      v16bf bg = cat8(*(const v8bf*)brg, *(const v8bf*)(brg + 8));
      #pragma unroll
      for (int mt = 0; mt < 2; ++mt) {
        accN[mt][nt] = wmma_bf16(af[mt], bn, accN[mt][nt]);
        accG[mt][nt] = wmma_bf16(af[mt], bg, accG[mt][nt]);
      }
    }

    if (kt + 1 < NK) {
      storeS(buf ^ 1);
      __syncthreads();
      buf ^= 1;
    }
  }

  #pragma unroll
  for (int nt = 0; nt < 2; ++nt) {
    int n = n0 + wn * 32 + nt * 16 + lh;
    float bN = bias[n];
    float bG = bias[n + H_];
    #pragma unroll
    for (int mt = 0; mt < 2; ++mt) {
      #pragma unroll
      for (int r = 0; r < 8; ++r) {
        int m = m0 + wm * 32 + mt * 16 + r + ls * 8;
        float nl = accN[mt][nt][r] + bN;
        float gt = accG[mt][nt][r] + bG;
        float g  = 1.0f / (1.0f + __expf(-gt));
        float xo = X[(long)m * H_ + n];
        float y  = g * xo + (1.0f - g) * fmaxf(nl, 0.0f);
        Out1[(long)m * H_ + n] = y;
        if (Out2) Out2[(long)m * ldo2 + n] = y;
      }
    }
  }
}

// ---------------------------------------------------------------------------
// Host orchestration
// ---------------------------------------------------------------------------
extern "C" void kernel_launch(void* const* d_in, const int* in_sizes, int n_in,
                              void* d_out, int out_size, void* d_ws, size_t ws_size,
                              hipStream_t stream) {
  (void)in_sizes; (void)n_in; (void)out_size; (void)ws_size;

  const float* inputs   = (const float*)d_in[0];
  const float* fwd_pad  = (const float*)d_in[1];
  const float* bwd_pad  = (const float*)d_in[2];
  const float* fwd_w    = (const float*)d_in[3];
  const float* fwd_b    = (const float*)d_in[4];
  const float* bwd_w    = (const float*)d_in[5];
  const float* bwd_b    = (const float*)d_in[6];
  const float* fwd_hw_w = (const float*)d_in[7];
  const float* fwd_hw_b = (const float*)d_in[8];
  const float* bwd_hw_w = (const float*)d_in[9];
  const float* bwd_hw_b = (const float*)d_in[10];
  float* out = (float*)d_out;

  // Workspace carve (256B aligned slots)
  char* ws = (char*)d_ws;
  size_t off = 0;
  auto carve = [&](size_t bytes) -> char* {
    char* p = ws + off;
    off = (off + bytes + 255) & ~(size_t)255;
    return p;
  };
  const long nW  = (long)L_ * H_ * IN_;            // conv weights per direction
  const long nHW = (long)L_ * NHW_ * 2 * H_ * H_;  // highway weights per direction
  bf16_t* wF   = (bf16_t*)carve(nW * 2);
  bf16_t* wB   = (bf16_t*)carve(nW * 2);
  bf16_t* hwF  = (bf16_t*)carve(nHW * 2);
  bf16_t* hwB  = (bf16_t*)carve(nHW * 2);
  float* padF  = (float*)carve((long)B_ * SP_ * H_ * 4);
  float* padBk = (float*)carve((long)B_ * SP_ * H_ * 4);
  float* bufAf = (float*)carve((long)B_ * S_ * H_ * 4);
  float* bufBf = (float*)carve((long)B_ * S_ * H_ * 4);
  float* bufAb = (float*)carve((long)B_ * S_ * H_ * 4);
  float* bufBb = (float*)carve((long)B_ * S_ * H_ * 4);

  // Convert all weights to bf16 once per launch (deterministic, no caching)
  cvt_bf16_kernel<<<dim3((unsigned)((nW  + 255) / 256)), 256, 0, stream>>>(fwd_w,    wF,  nW);
  cvt_bf16_kernel<<<dim3((unsigned)((nW  + 255) / 256)), 256, 0, stream>>>(bwd_w,    wB,  nW);
  cvt_bf16_kernel<<<dim3((unsigned)((nHW + 255) / 256)), 256, 0, stream>>>(fwd_hw_w, hwF, nHW);
  cvt_bf16_kernel<<<dim3((unsigned)((nHW + 255) / 256)), 256, 0, stream>>>(bwd_hw_w, hwB, nHW);

  const float* xf = inputs;
  const float* xb = inputs;
  const long padn = (long)B_ * SP_ * H_;
  const dim3 gpad((unsigned)((padn + 255) / 256));
  const dim3 gconv(H_ / 128, S_ / 128, B_);          // (4, 4, 32)
  const dim3 ghw(H_ / 64, (B_ * S_) / 128, 1);       // (8, 128)

  for (int i = 0; i < L_; ++i) {
    // Padded sequences (both directions padded with fwd_pad front / bwd_pad back)
    build_padded_kernel<<<gpad, 256, 0, stream>>>(xf, fwd_pad + i * W_ * H_, bwd_pad + i * W_ * H_, padF);
    build_padded_kernel<<<gpad, 256, 0, stream>>>(xb, fwd_pad + i * W_ * H_, bwd_pad + i * W_ * H_, padBk);

    // Conv GEMMs: forward window = offset 0, backward window = offset W*H
    conv_gemm_relu_kernel<<<gconv, 256, 0, stream>>>(
        padF,  (long)SP_ * H_, 0,              H_,
        wF + (long)i * H_ * IN_, IN_, fwd_b + i * H_,
        bufAf, (long)S_ * H_, H_, IN_);
    conv_gemm_relu_kernel<<<gconv, 256, 0, stream>>>(
        padBk, (long)SP_ * H_, (long)W_ * H_,  H_,
        wB + (long)i * H_ * IN_, IN_, bwd_b + i * H_,
        bufAb, (long)S_ * H_, H_, IN_);

    // Highway stages (fully fused gate + residual; last stage also writes d_out)
    for (int j = 0; j < NHW_; ++j) {
      const float* inF = (j == 0) ? bufAf : bufBf;
      float*       oF  = (j == 0) ? bufBf : bufAf;
      const float* inB = (j == 0) ? bufAb : bufBb;
      float*       oB  = (j == 0) ? bufBb : bufAb;
      float* o2F = (j == NHW_ - 1) ? (out + (long)i * B_ * S_ * 2 * H_)       : nullptr;
      float* o2B = (j == NHW_ - 1) ? (out + (long)i * B_ * S_ * 2 * H_ + H_)  : nullptr;
      highway_gemm_kernel<<<ghw, 256, 0, stream>>>(
          inF, hwF + ((long)i * NHW_ + j) * 2 * H_ * H_,
          fwd_hw_b + ((long)i * NHW_ + j) * 2 * H_, oF, o2F, 2 * H_);
      highway_gemm_kernel<<<ghw, 256, 0, stream>>>(
          inB, hwB + ((long)i * NHW_ + j) * 2 * H_ * H_,
          bwd_hw_b + ((long)i * NHW_ + j) * 2 * H_, oB, o2B, 2 * H_);
    }
    xf = bufAf;  // j=1 wrote back into bufA*
    xb = bufAb;
  }
}
